// Fuser_V3_85555748536546
// MI455X (gfx1250) — matrix operand order repaired and checked by reading, not compiled, not run on hardware
//
#include <hip/hip_runtime.h>
#include <hip/hip_bf16.h>

typedef __attribute__((ext_vector_type(16))) _Float16 v16h;
typedef __attribute__((ext_vector_type(8)))  _Float16 v8h;
typedef __attribute__((ext_vector_type(4)))  _Float16 h4;
typedef __attribute__((ext_vector_type(8)))  float    v8f;
typedef __attribute__((ext_vector_type(4)))  float    f4;
typedef __attribute__((ext_vector_type(4)))  int      v4i;

#define S_LEN   4096
#define D_MODEL 128
#define KBLK    32
#define WAVES   4     // 4 waves * 16 queries = 64 queries per workgroup

#if __has_builtin(__builtin_amdgcn_global_load_async_to_lds_b128)
#define HAVE_ASYNC_LDS 1
#else
#define HAVE_ASYNC_LDS 0
#endif

__device__ __forceinline__ v8f wmma_f16(v16h a, v16h b, v8f c) {
  return __builtin_amdgcn_wmma_f32_16x16x32_f16(false, a, false, b, (short)0, c,
                                                false, false);
}

// One 16-byte async global->LDS copy (ASYNCcnt-tracked), manual fallback.
__device__ __forceinline__ void copy16_g2lds(const _Float16* gsrc, _Float16* ldst) {
#if HAVE_ASYNC_LDS
  __builtin_amdgcn_global_load_async_to_lds_b128(
      (__attribute__((address_space(1))) v4i*)gsrc,
      (__attribute__((address_space(3))) v4i*)ldst,
      0, 0);
#else
  *(v8h*)ldst = *(const v8h*)gsrc;
#endif
}

__device__ __forceinline__ void wait_async_copies() {
#if HAVE_ASYNC_LDS
#if __has_builtin(__builtin_amdgcn_s_wait_asynccnt)
  __builtin_amdgcn_s_wait_asynccnt(0);
#else
  asm volatile("s_wait_asynccnt 0x0" ::: "memory");
#endif
#endif
}

// ---------------------------------------------------------------------------
// Kernel 0: one-time fp32 -> fp16 conversion of x into natural [b][s][d] and
// transposed [b][d][s] copies.
// ---------------------------------------------------------------------------
__global__ __launch_bounds__(256)
void cvt_kernel(const float* __restrict__ x,
                _Float16* __restrict__ xh,
                _Float16* __restrict__ xhT) {
  const size_t i = (size_t)blockIdx.x * 256 + threadIdx.x;   // one float4 each
  const f4 v = ((const f4*)x)[i];
  h4 h;
#pragma unroll
  for (int j = 0; j < 4; ++j) h[j] = (_Float16)v[j];
  ((h4*)xh)[i] = h;

  const size_t f   = i * 4;
  const int    b   = (int)(f / ((size_t)S_LEN * D_MODEL));
  const size_t rem = f - (size_t)b * S_LEN * D_MODEL;
  const int    s   = (int)(rem / D_MODEL);
  const int    d   = (int)(rem % D_MODEL);
#pragma unroll
  for (int j = 0; j < 4; ++j)
    xhT[((size_t)b * D_MODEL + d + j) * S_LEN + s] = h[j];
}

// ---------------------------------------------------------------------------
// Kernel 1: transposed flash attention (S^T = K*Q^T, O^T += V^T*P^T), f16
// WMMA w/ f32 accumulation, streaming softmax, double-buffered async LDS.
// ---------------------------------------------------------------------------
__global__ __launch_bounds__(WAVES * 32)
void fa_kernel(const _Float16* __restrict__ xh,   // [B][S][D] f16
               const _Float16* __restrict__ xhT,  // [B][D][S] f16
               const float* __restrict__ att_w,
               const float* __restrict__ att_b,
               float* __restrict__ hbuf,
               float* __restrict__ psbuf) {
  // double-buffered tiles: natural [key][d] and transposed [d][key], f16
  __shared__ _Float16 tileA[2][KBLK * D_MODEL];   // 2 x 8 KB
  __shared__ _Float16 tileT[2][D_MODEL * KBLK];   // 2 x 8 KB

  const int tid  = threadIdx.x;
  const int lane = tid & 31;
  const int wave = tid >> 5;
  const int lo   = lane & 15;
  const int half = lane >> 4;
  const int b    = blockIdx.y;
  const int qb   = blockIdx.x * (WAVES * 16) + wave * 16;

  const _Float16* xb  = xh  + (size_t)b * S_LEN * D_MODEL;
  const _Float16* xbT = xhT + (size_t)b * D_MODEL * S_LEN;
  const float scale = 0.08838834764831845f;   // 1/sqrt(128)

  // ---- Q fragments in B-matrix layout: lane -> query col, elem e -> K ----
  v16h Bq[4];
  {
    const _Float16* qrow = xb + (size_t)(qb + lo) * D_MODEL;
#pragma unroll
    for (int c = 0; c < 4; ++c) {
      const v8h q0 = *(const v8h*)(qrow + c * 32 + half * 16);
      const v8h q1 = *(const v8h*)(qrow + c * 32 + half * 16 + 8);
#pragma unroll
      for (int e = 0; e < 8; ++e) { Bq[c][e] = q0[e]; Bq[c][8 + e] = q1[e]; }
    }
  }

  v8f O[8];
#pragma unroll
  for (int t = 0; t < 8; ++t)
#pragma unroll
    for (int r = 0; r < 8; ++r) O[t][r] = 0.0f;

  float M = -3.402823466e38f;
  float L = 0.0f;

  // stage one tile (32 keys x 128 d, both layouts; 128 thr x 8 x 16 B)
  auto stage_tile = [&](int buf, int kb) {
#pragma unroll
    for (int i = 0; i < 4; ++i) {            // natural: contiguous 8 KB
      const int idx = tid + i * 128;
      copy16_g2lds(xb + (size_t)kb * D_MODEL + idx * 8, &tileA[buf][idx * 8]);
    }
#pragma unroll
    for (int i = 0; i < 4; ++i) {            // transposed: 128 rows x 32 keys
      const int idx  = tid + i * 128;
      const int d    = idx >> 2;
      const int part = idx & 3;
      copy16_g2lds(xbT + (size_t)d * S_LEN + kb + part * 8,
                   &tileT[buf][d * KBLK + part * 8]);
    }
  };

  stage_tile(0, 0);
  wait_async_copies();
  __syncthreads();

  int ib = 0;
  for (int kb = 0; kb < S_LEN; kb += KBLK, ib ^= 1) {
    // ---- kick off async copies for the NEXT tile (overlaps compute) ----
    if (kb + KBLK < S_LEN) stage_tile(ib ^ 1, kb + KBLK);
    if (kb + 2 * KBLK < S_LEN)
      __builtin_prefetch(xb + (size_t)(kb + 2 * KBLK) * D_MODEL + tid * 32, 0, 0);

    const _Float16* curA = tileA[ib];
    const _Float16* curT = tileT[ib];

    // ---- S^T tiles (rows = keys, cols = queries), ping-pong fragments ----
    v8f s0, s1;
#pragma unroll
    for (int r = 0; r < 8; ++r) { s0[r] = 0.0f; s1[r] = 0.0f; }

    auto loadA = [&](int f) -> v16h {        // f: kh = f&1, c = f>>1
      const _Float16* p =
          curA + ((f & 1) * 16 + lo) * D_MODEL + (f >> 1) * 32 + half * 16;
      const v8h a0 = *(const v8h*)p, a1 = *(const v8h*)(p + 8);
      v16h a;
#pragma unroll
      for (int e = 0; e < 8; ++e) { a[e] = a0[e]; a[8 + e] = a1[e]; }
      return a;
    };

    {
      v16h fa = loadA(0);
      v16h fb = loadA(1);
#pragma unroll
      for (int f = 0; f < 8; ++f) {
        v16h nxt = (f < 6) ? loadA(f + 2) : fb;   // issue ahead of the WMMA
        if (f & 1) s1 = wmma_f16(fa, Bq[f >> 1], s1);
        else       s0 = wmma_f16(fa, Bq[f >> 1], s0);
        fa = fb;
        fb = nxt;
      }
    }

    // ---- streaming softmax over this 32-key block ----
#pragma unroll
    for (int r = 0; r < 8; ++r) { s0[r] *= scale; s1[r] *= scale; }

    float mb = -3.402823466e38f;
#pragma unroll
    for (int r = 0; r < 8; ++r) mb = fmaxf(mb, fmaxf(s0[r], s1[r]));
    mb = fmaxf(mb, __shfl_xor(mb, 16));

    float Mn    = fmaxf(M, mb);
    float alpha = __expf(M - Mn);
#pragma unroll
    for (int t = 0; t < 8; ++t)
#pragma unroll
      for (int r = 0; r < 8; ++r) O[t][r] *= alpha;

    float p0[8], p1[8], ps = 0.0f;
#pragma unroll
    for (int r = 0; r < 8; ++r) {
      p0[r] = __expf(s0[r] - Mn);
      p1[r] = __expf(s1[r] - Mn);
      ps += p0[r] + p1[r];
    }
    ps += __shfl_xor(ps, 16);
    L = L * alpha + ps;
    M = Mn;

    // ---- P^T: C layout -> B layout via one shfl_xor(16) per VGPR ----
    v16h bp;
#pragma unroll
    for (int r = 0; r < 8; ++r) {
      float o0 = __shfl_xor(p0[r], 16);
      float o1 = __shfl_xor(p1[r], 16);
      bp[r]     = (_Float16)(half ? o1    : p0[r]);
      bp[8 + r] = (_Float16)(half ? p1[r] : o0);
    }

    // ---- O^T += V^T * P^T, ping-pong V fragments ----
    auto loadV = [&](int t) -> v16h {
      const _Float16* p = curT + (16 * t + lo) * KBLK + half * 16;
      const v8h a0 = *(const v8h*)p, a1 = *(const v8h*)(p + 8);
      v16h a;
#pragma unroll
      for (int e = 0; e < 8; ++e) { a[e] = a0[e]; a[8 + e] = a1[e]; }
      return a;
    };

    {
      v16h va = loadV(0);
#pragma unroll
      for (int t = 0; t < 8; ++t) {
        v16h nv = (t < 7) ? loadV(t + 1) : va;    // issue ahead of the WMMA
        O[t] = wmma_f16(va, bp, O[t]);
        va = nv;
      }
    }

    wait_async_copies();   // next tile landed (own copies)
    __syncthreads();       // everyone's copies landed, reads of cur done
  }

  // ---- epilogue: h = O/L (f32), pool score = h.w + b ----
  const float inv = 1.0f / L;
  const int   q   = qb + lo;
  float* hrow = hbuf + ((size_t)b * S_LEN + q) * D_MODEL;

  float pp = 0.0f;
#pragma unroll
  for (int t = 0; t < 8; ++t) {
    const int dbase = 16 * t + half * 8;
    f4 v0, v1;
#pragma unroll
    for (int r = 0; r < 4; ++r) {
      v0[r] = O[t][r] * inv;
      v1[r] = O[t][4 + r] * inv;
    }
    *(f4*)(hrow + dbase)     = v0;
    *(f4*)(hrow + dbase + 4) = v1;
#pragma unroll
    for (int r = 0; r < 8; ++r)
      pp += O[t][r] * inv * att_w[dbase + r];
  }
  pp += __shfl_xor(pp, 16);
  pp += att_b[0];
  if (half == 0) psbuf[b * S_LEN + q] = pp;
}

// ---------------------------------------------------------------------------
// Kernel 2: attention pooling. One block per batch, thread t owns feature t.
// ---------------------------------------------------------------------------
__global__ __launch_bounds__(D_MODEL)
void pool_kernel(const float* __restrict__ hbuf,
                 const float* __restrict__ psbuf,
                 float* __restrict__ out) {
  __shared__ float red[D_MODEL];
  const int b = blockIdx.x;
  const int t = threadIdx.x;
  const float* s = psbuf + (size_t)b * S_LEN;

  float m = -3.402823466e38f;
  for (int q = t; q < S_LEN; q += D_MODEL) m = fmaxf(m, s[q]);
  red[t] = m;
  __syncthreads();
  for (int off = D_MODEL / 2; off > 0; off >>= 1) {
    if (t < off) red[t] = fmaxf(red[t], red[t + off]);
    __syncthreads();
  }
  m = red[0];
  __syncthreads();

  float z = 0.0f;
  for (int q = t; q < S_LEN; q += D_MODEL) z += __expf(s[q] - m);
  red[t] = z;
  __syncthreads();
  for (int off = D_MODEL / 2; off > 0; off >>= 1) {
    if (t < off) red[t] += red[t + off];
    __syncthreads();
  }
  z = red[0];
  __syncthreads();

  const float* hb = hbuf + (size_t)b * S_LEN * D_MODEL;
  float acc = 0.0f;
  for (int q = 0; q < S_LEN; ++q)
    acc += __expf(s[q] - m) * hb[(size_t)q * D_MODEL + t];
  out[b * D_MODEL + t] = acc / z;
}

// ---------------------------------------------------------------------------
extern "C" void kernel_launch(void* const* d_in, const int* in_sizes, int n_in,
                              void* d_out, int out_size, void* d_ws, size_t ws_size,
                              hipStream_t stream) {
  (void)in_sizes; (void)n_in; (void)out_size; (void)ws_size;
  const float* x     = (const float*)d_in[0];   // [4, 4096, 128] f32
  const float* att_w = (const float*)d_in[1];   // [128, 1] f32
  const float* att_b = (const float*)d_in[2];   // [1] f32
  float* out = (float*)d_out;                   // [4, 128] f32

  // workspace layout
  float*    hbuf  = (float*)d_ws;                                   // 8 MB
  float*    psbuf = hbuf + (size_t)4 * S_LEN * D_MODEL;             // 64 KB
  _Float16* xhb   = (_Float16*)(psbuf + (size_t)4 * S_LEN);         // 4 MB
  _Float16* xhTb  = xhb + (size_t)4 * S_LEN * D_MODEL;              // 4 MB

  const int n4 = 4 * S_LEN * D_MODEL / 4;   // float4 elements
  cvt_kernel<<<n4 / 256, 256, 0, stream>>>(x, xhb, xhTb);

  dim3 grid1(S_LEN / (WAVES * 16), 4);
  fa_kernel<<<grid1, WAVES * 32, 0, stream>>>(xhb, xhTb, att_w, att_b,
                                              hbuf, psbuf);
  pool_kernel<<<4, D_MODEL, 0, stream>>>(hbuf, psbuf, out);
}
